// MyModel_87196426043721
// MI455X (gfx1250) — matrix operand order repaired and checked by reading, not compiled, not run on hardware
//
#include <hip/hip_runtime.h>

// ---------------------------------------------------------------------------
// CDNA5 (gfx1250) WMMA types
// ---------------------------------------------------------------------------
typedef __attribute__((ext_vector_type(16))) __bf16 v16bf;
typedef __attribute__((ext_vector_type(8)))  __bf16 v8bf;   // 16 bytes
typedef __attribute__((ext_vector_type(4)))  __bf16 v4bf;   // 8 bytes
typedef __attribute__((ext_vector_type(8)))  float  v8f;

__device__ __forceinline__ v8f wmma_bf16(v16bf a, v16bf b, v8f c) {
  // D(16x16 f32) = A(16x32 bf16) x B(32x16 bf16) + C
  return __builtin_amdgcn_wmma_f32_16x16x32_bf16(
      /*neg_a=*/false, a, /*neg_b=*/false, b,
      /*c_mod=*/(short)0, c, /*reuse_a=*/false, /*reuse_b=*/false);
}

// A-fragment (16x32 bf16, M x K) from an LDS tile, per ISA 7.12.2.
// Per lane: two contiguous 8-element (16 B) chunks -> two ds_load_b128.
__device__ __forceinline__ v16bf load_a_frag(const __bf16* sm, int ldw,
                                             int colbase, int ktile, int lane) {
  const int h = (lane >> 4) & 1;
  const int m = lane & 15;
  const __bf16* row = sm + m * ldw + colbase + ktile * 32;
  v8bf lo = *(const v8bf*)(row + h * 8);
  v8bf hi = *(const v8bf*)(row + 16 + h * 8);
  return __builtin_shufflevector(lo, hi, 0, 1, 2, 3, 4, 5, 6, 7,
                                 8, 9, 10, 11, 12, 13, 14, 15);
}

// B-fragment (32x16 bf16, K x N) from PRE-PACKED, ZERO-PADDED bf16 weights
// Wp[noutp][ldk].  Two 8-bf16 chunks per lane (k=8h.., 16+8h..) -> two global_load_b128.
__device__ __forceinline__ v16bf load_b_frag(const __bf16* __restrict__ Wp,
                                             int ldk, int ntile, int ktile,
                                             int lane) {
  const int h = (lane >> 4) & 1;
  const int n = lane & 15;
  const __bf16* base = Wp + (ntile * 16 + n) * ldk + ktile * 32 + h * 8;
  v8bf lo = *(const v8bf*)(base);
  v8bf hi = *(const v8bf*)(base + 16);
  return __builtin_shufflevector(lo, hi, 0, 1, 2, 3, 4, 5, 6, 7,
                                 8, 9, 10, 11, 12, 13, 14, 15);
}

// Issue NT B-fragment load pairs as one batch (one s_clause of b128 loads),
// then consume with the WMMA chain -> staggered waitcnt instead of a full
// stop per tile.
template <int NT>
__device__ __forceinline__ void gemm_chunk(v16bf a,
                                           const __bf16* __restrict__ Wp,
                                           int ldk, int ntbase, int kt,
                                           int lane, v8f* acc) {
  v16bf bf[NT];
#pragma unroll
  for (int j = 0; j < NT; ++j) bf[j] = load_b_frag(Wp, ldk, ntbase + j, kt, lane);
#pragma unroll
  for (int j = 0; j < NT; ++j) acc[ntbase + j] = wmma_bf16(a, bf[j], acc[ntbase + j]);
}

// ---------------------------------------------------------------------------
// Weight packing: f32 [nout,kdim] -> bf16 [noutp,kp], zero-padded.
// ---------------------------------------------------------------------------
__global__ void k_pack_w(const float* __restrict__ W, int nout, int kdim,
                         __bf16* __restrict__ Wp, int noutp, int kp) {
  for (int i = blockIdx.x * blockDim.x + threadIdx.x; i < noutp * kp;
       i += gridDim.x * blockDim.x) {
    const int n = i / kp, k = i % kp;
    const float v = (n < nout && k < kdim) ? W[n * kdim + k] : 0.0f;
    Wp[i] = (__bf16)v;
  }
}

__global__ void k_pack_bias(const float* __restrict__ b, int n,
                            float* __restrict__ bp, int np) {
  const int i = blockIdx.x * blockDim.x + threadIdx.x;
  if (i < np) bp[i] = (i < n) ? b[i] : 0.0f;
}

// ---------------------------------------------------------------------------
// Fused embed MLP chain:  [rows,344] -> [rows,32]   (rows % 64 == 0)
// One 16-row tile per wave, 4 waves / block.  bf16 WMMA, f32 accumulate;
// intermediates round-trip LDS in bf16 (DS ops are in-order per wave and
// each wave owns its LDS slice -> no barriers).
// ---------------------------------------------------------------------------
#define EMB_LDW 352

__global__ __launch_bounds__(128) void k_embed(
    const float* __restrict__ in, int rows,
    const __bf16* __restrict__ W1p, const float* __restrict__ b1p,  // 176x320
    const __bf16* __restrict__ W2p, const float* __restrict__ b2p,  // 96x192
    const __bf16* __restrict__ W3p, const float* __restrict__ b3,   // 32x128
    const __bf16* __restrict__ W4p, const float* __restrict__ b4,   // 3x32x32
    float* __restrict__ out)                                        // rows x 32
{
  __shared__ __bf16 smem[4][16][EMB_LDW];
  const int wave = threadIdx.x >> 5;
  const int lane = threadIdx.x & 31;
  const int tileRow = (blockIdx.x * 4 + wave) * 16;
  if (tileRow >= rows) return;

  __bf16* sm = &smem[wave][0][0];

  // Stage 16 input rows: float4 loads -> packed v4bf LDS stores (no guards).
  for (int idx = lane; idx < 16 * 86; idx += 32) {
    const int r = idx / 86, c4 = idx % 86;
    const float4 f = ((const float4*)(in + (long)(tileRow + r) * 344))[c4];
    v4bf o;
    o[0] = (__bf16)f.x; o[1] = (__bf16)f.y;
    o[2] = (__bf16)f.z; o[3] = (__bf16)f.w;
    *(v4bf*)(sm + r * EMB_LDW + c4 * 4) = o;
  }
  v4bf z4;
  z4[0] = (__bf16)0.f; z4[1] = (__bf16)0.f;
  z4[2] = (__bf16)0.f; z4[3] = (__bf16)0.f;
  for (int idx = lane; idx < 16 * 2; idx += 32) {        // zero cols 344..352
    const int r = idx / 2, c = idx % 2;
    *(v4bf*)(sm + r * EMB_LDW + 344 + c * 4) = z4;
  }

  const v8f vzero = {0.f, 0.f, 0.f, 0.f, 0.f, 0.f, 0.f, 0.f};
  const int h = (lane >> 4) & 1;
  const int n = lane & 15;

  // ---- GEMM1: out1[16,176] = A[:,24:344] @ W1p^T   (K=320) ----
  v8f acc1[11];
#pragma unroll
  for (int t = 0; t < 11; ++t) acc1[t] = vzero;
#pragma unroll
  for (int kt = 0; kt < 10; ++kt) {
    v16bf a = load_a_frag(sm, EMB_LDW, 24, kt, lane);
    gemm_chunk<6>(a, W1p, 320, 0, kt, lane, acc1);
    gemm_chunk<5>(a, W1p, 320, 6, kt, lane, acc1);
  }
  // Bias + store out1 (bf16) to LDS cols [24,200); zero-pad [200,216).
#pragma unroll
  for (int nt = 0; nt < 11; ++nt) {
    const int col  = nt * 16 + n;
    const float bb = b1p[col];
#pragma unroll
    for (int v = 0; v < 8; ++v) {
      const int m = v + 8 * h;
      sm[m * EMB_LDW + 24 + col] = (__bf16)(acc1[nt][v] + bb);
    }
  }
  for (int idx = lane; idx < 16 * 4; idx += 32) {
    const int r = idx / 4, c = idx % 4;
    *(v4bf*)(sm + r * EMB_LDW + 200 + c * 4) = z4;
  }

  // ---- GEMM2: out2[16,96] = out1 @ W2p^T   (K=192) ----
  v8f acc2[6];
#pragma unroll
  for (int t = 0; t < 6; ++t) acc2[t] = vzero;
#pragma unroll
  for (int kt = 0; kt < 6; ++kt) {
    v16bf a = load_a_frag(sm, EMB_LDW, 24, kt, lane);
    gemm_chunk<6>(a, W2p, 192, 0, kt, lane, acc2);
  }
  // x110 = [cols 0..24 original | out2 at cols 24..120); pad [120,128).
#pragma unroll
  for (int nt = 0; nt < 6; ++nt) {
    const int col  = nt * 16 + n;
    const float bb = b2p[col];
#pragma unroll
    for (int v = 0; v < 8; ++v) {
      const int m = v + 8 * h;
      sm[m * EMB_LDW + 24 + col] = (__bf16)(acc2[nt][v] + bb);
    }
  }
  for (int idx = lane; idx < 16 * 2; idx += 32) {
    const int r = idx / 2, c = idx % 2;
    *(v4bf*)(sm + r * EMB_LDW + 120 + c * 4) = z4;
  }

  // ---- GEMM3: x32[16,32] = x110 @ W3p^T   (K=128) ----
  v8f acc3[2];
  acc3[0] = vzero; acc3[1] = vzero;
#pragma unroll
  for (int kt = 0; kt < 4; ++kt) {
    v16bf a = load_a_frag(sm, EMB_LDW, 0, kt, lane);
    gemm_chunk<2>(a, W3p, 128, 0, kt, lane, acc3);
  }
#pragma unroll
  for (int nt = 0; nt < 2; ++nt) {
    const int col = nt * 16 + n;
#pragma unroll
    for (int v = 0; v < 8; ++v) {
      const int m = v + 8 * h;
      sm[m * EMB_LDW + col] = (__bf16)(acc3[nt][v] + b3[col]);
    }
  }

  // ---- 3 x (x32 = x32 @ W4[l]^T + b4[l]) ----
#pragma unroll
  for (int l = 0; l < 3; ++l) {
    const __bf16* Wl = W4p + l * 32 * 32;
    const float*  bl = b4 + l * 32;
    v16bf a = load_a_frag(sm, EMB_LDW, 0, 0, lane);
    v8f accl[2];
    accl[0] = vzero; accl[1] = vzero;
    gemm_chunk<2>(a, Wl, 32, 0, 0, lane, accl);
    if (l < 2) {
#pragma unroll
      for (int v = 0; v < 8; ++v) {
        const int m = v + 8 * h;
        sm[m * EMB_LDW + n]      = (__bf16)(accl[0][v] + bl[n]);
        sm[m * EMB_LDW + 16 + n] = (__bf16)(accl[1][v] + bl[16 + n]);
      }
    } else {
#pragma unroll
      for (int v = 0; v < 8; ++v) {
        const int m  = v + 8 * h;
        const long gr = tileRow + m;
        out[gr * 32 + n]      = accl[0][v] + bl[n];
        out[gr * 32 + 16 + n] = accl[1][v] + bl[16 + n];
      }
    }
  }
}

// ---------------------------------------------------------------------------
// GAT projection:  xp[rows,32] = x[rows,32] @ Wp^T   (rows % 64 == 0) — WMMA
// ---------------------------------------------------------------------------
__global__ __launch_bounds__(128) void k_proj(const float* __restrict__ x,
                                              int rows,
                                              const __bf16* __restrict__ Wp,
                                              float* __restrict__ xp) {
  __shared__ __bf16 smem[4][16][32];
  const int wave = threadIdx.x >> 5;
  const int lane = threadIdx.x & 31;
  const int tileRow = (blockIdx.x * 4 + wave) * 16;
  if (tileRow >= rows) return;
  __bf16* sm = &smem[wave][0][0];
  for (int idx = lane; idx < 16 * 8; idx += 32) {
    const int r = idx / 8, c4 = idx % 8;
    const float4 f = ((const float4*)(x + (long)(tileRow + r) * 32))[c4];
    v4bf o;
    o[0] = (__bf16)f.x; o[1] = (__bf16)f.y;
    o[2] = (__bf16)f.z; o[3] = (__bf16)f.w;
    *(v4bf*)(sm + r * 32 + c4 * 4) = o;
  }
  const v8f vzero = {0.f, 0.f, 0.f, 0.f, 0.f, 0.f, 0.f, 0.f};
  v16bf a = load_a_frag(sm, 32, 0, 0, lane);
  v8f acc[2];
  acc[0] = vzero; acc[1] = vzero;
  gemm_chunk<2>(a, Wp, 32, 0, 0, lane, acc);
  const int h = (lane >> 4) & 1;
  const int n = lane & 15;
#pragma unroll
  for (int v = 0; v < 8; ++v) {
    const long gr = tileRow + v + 8 * h;
    xp[gr * 32 + n]      = acc[0][v];
    xp[gr * 32 + 16 + n] = acc[1][v];
  }
}

// attention coefficients: as_[n,h] = sum_d xp[n,h,d]*a_src[h,d]  (same for dst)
template <int H, int D>
__global__ void k_coef(const float* __restrict__ xp, int n,
                       const float* __restrict__ asrc,
                       const float* __restrict__ adst,
                       float* __restrict__ as_, float* __restrict__ ad_) {
  const int i = blockIdx.x * blockDim.x + threadIdx.x;
  if (i >= n * H) return;
  const int node = i / H, hh = i % H;
  float sa = 0.f, sd = 0.f;
#pragma unroll
  for (int d = 0; d < D; ++d) {
    const float v = xp[(long)node * 32 + hh * D + d];
    sa += v * asrc[hh * D + d];
    sd += v * adst[hh * D + d];
  }
  as_[i] = sa;
  ad_[i] = sd;
}

__global__ void k_fill(float* __restrict__ p, float v, int n) {
  for (int i = blockIdx.x * blockDim.x + threadIdx.x; i < n;
       i += gridDim.x * blockDim.x)
    p[i] = v;
}

__device__ __forceinline__ void atomicMaxF(float* addr, float val) {
  int* ia  = (int*)addr;
  int  old = __float_as_int(*addr);
  while (__int_as_float(old) < val) {
    const int prev = atomicCAS(ia, old, __float_as_int(val));
    if (prev == old) break;
    old = prev;
  }
}

__device__ __forceinline__ void edge_sd(const int* ei, int E, int e,
                                        int& s, int& d) {
  if (e < E) { s = ei[e]; d = ei[E + e]; }
  else       { s = e - E; d = e - E; }          // self-loop
}

template <int H>
__global__ void k_edge_max(const int* __restrict__ ei, int E, int N,
                           const float* __restrict__ as_,
                           const float* __restrict__ ad_,
                           float* __restrict__ mmax) {
  const int idx = blockIdx.x * blockDim.x + threadIdx.x;
  const int tot = (E + N) * H;
  if (idx >= tot) return;
  const int e = idx / H, hh = idx % H;
  int s, d; edge_sd(ei, E, e, s, d);
  float v = as_[s * H + hh] + ad_[d * H + hh];
  v = (v > 0.f) ? v : 0.2f * v;
  atomicMaxF(&mmax[d * H + hh], v);
}

template <int H>
__global__ void k_edge_sum(const int* __restrict__ ei, int E, int N,
                           const float* __restrict__ as_,
                           const float* __restrict__ ad_,
                           const float* __restrict__ mmax,
                           float* __restrict__ ssum) {
  const int idx = blockIdx.x * blockDim.x + threadIdx.x;
  const int tot = (E + N) * H;
  if (idx >= tot) return;
  const int e = idx / H, hh = idx % H;
  int s, d; edge_sd(ei, E, e, s, d);
  float v = as_[s * H + hh] + ad_[d * H + hh];
  v = (v > 0.f) ? v : 0.2f * v;
  atomicAdd(&ssum[d * H + hh], __expf(v - mmax[d * H + hh]));
}

template <int H, int D>
__global__ void k_edge_agg(const int* __restrict__ ei, int E, int N,
                           const float* __restrict__ as_,
                           const float* __restrict__ ad_,
                           const float* __restrict__ mmax,
                           const float* __restrict__ ssum,
                           const float* __restrict__ xp,
                           float* __restrict__ agg) {
  const int idx = blockIdx.x * blockDim.x + threadIdx.x;
  const int tot = (E + N) * H;
  if (idx >= tot) return;
  const int e = idx / H, hh = idx % H;
  __builtin_prefetch(&ei[(e + 2048 < E) ? e + 2048 : e], 0, 1);  // edge stream
  int s, d; edge_sd(ei, E, e, s, d);
  float v = as_[s * H + hh] + ad_[d * H + hh];
  v = (v > 0.f) ? v : 0.2f * v;
  const float alpha = __expf(v - mmax[d * H + hh]) / (ssum[d * H + hh] + 1e-16f);
#pragma unroll
  for (int dd = 0; dd < D; ++dd)
    atomicAdd(&agg[(long)d * 32 + hh * D + dd],
              alpha * xp[(long)s * 32 + hh * D + dd]);
}

__global__ void k_bias_act(const float* __restrict__ agg,
                           const float* __restrict__ b,
                           float* __restrict__ out, int n, int elu) {
  const int i = blockIdx.x * blockDim.x + threadIdx.x;
  if (i >= n) return;
  float v = agg[i] + b[i & 31];                 // C == 32
  if (elu) v = (v > 0.f) ? v : (__expf(v) - 1.0f);
  out[i] = v;
}

// (sum, sum_sq) reduction into red[0],red[1] via block-level LDS tree + atomics
__global__ __launch_bounds__(256) void k_redsum(const float* __restrict__ x,
                                                int n, float* __restrict__ red) {
  __shared__ float s1[256], s2[256];
  float a = 0.f, b = 0.f;
  for (int i = blockIdx.x * blockDim.x + threadIdx.x; i < n;
       i += gridDim.x * blockDim.x) {
    const float v = x[i];
    a += v; b += v * v;
  }
  s1[threadIdx.x] = a; s2[threadIdx.x] = b;
  __syncthreads();
  for (int off = 128; off > 0; off >>= 1) {
    if ((int)threadIdx.x < off) {
      s1[threadIdx.x] += s1[threadIdx.x + off];
      s2[threadIdx.x] += s2[threadIdx.x + off];
    }
    __syncthreads();
  }
  if (threadIdx.x == 0) { atomicAdd(&red[0], s1[0]); atomicAdd(&red[1], s2[0]); }
}

// Per-graph scoring: normalize x,ys with whole-tensor stats, A = xb @ ys^T,
// max over fragments, sigmoid(p-3).  One block per graph.
__global__ __launch_bounds__(256) void k_score(const float* __restrict__ x,
                                               const float* __restrict__ ys,
                                               const float* __restrict__ red,
                                               float* __restrict__ out,
                                               int npb, int FF, float cntx,
                                               float cnty) {
  __shared__ float ly[40 * 32];
  const int b  = blockIdx.x;
  const float mx = red[0] / cntx;
  const float ix = rsqrtf(red[1] / cntx - mx * mx + 1e-5f);
  const float my = red[2] / cnty;
  const float iy = rsqrtf(red[3] / cnty - my * my + 1e-5f);
  for (int i = threadIdx.x; i < FF * 32; i += 256)
    ly[i] = (ys[(long)b * FF * 32 + i] - my) * iy;
  __syncthreads();
  for (int node = threadIdx.x; node < npb; node += 256) {
    const long gn = (long)b * npb + node;
    float xr[32];
#pragma unroll
    for (int c = 0; c < 32; ++c) xr[c] = (x[gn * 32 + c] - mx) * ix;
    float p = -3.0e38f;
    for (int f = 0; f < FF; ++f) {
      float dot = 0.f;
#pragma unroll
      for (int c = 0; c < 32; ++c) dot += xr[c] * ly[f * 32 + c];
      p = fmaxf(p, dot);
    }
    out[gn] = 1.0f / (1.0f + __expf(-(p - 3.0f)));
  }
}

// ---------------------------------------------------------------------------
extern "C" void kernel_launch(void* const* d_in, const int* in_sizes, int n_in,
                              void* d_out, int out_size, void* d_ws,
                              size_t ws_size, hipStream_t stream) {
  (void)in_sizes; (void)n_in; (void)out_size; (void)ws_size;
  const float* frag = (const float*)d_in[0];
  const float* node = (const float*)d_in[1];
  const int*   ei   = (const int*)d_in[3];           // [2,E]; d_in[2] unused
  const float* ce1W = (const float*)d_in[4];
  const float* ce1b = (const float*)d_in[5];
  const float* ce2W = (const float*)d_in[6];
  const float* ce2b = (const float*)d_in[7];
  const float* lc0W = (const float*)d_in[8];
  const float* lc0b = (const float*)d_in[9];
  const float* lcW  = (const float*)d_in[10];
  const float* lcb  = (const float*)d_in[11];
  const float* g1W  = (const float*)d_in[12];
  const float* g1as = (const float*)d_in[13];
  const float* g1ad = (const float*)d_in[14];
  const float* g1b  = (const float*)d_in[15];
  const float* g2W  = (const float*)d_in[16];
  const float* g2as = (const float*)d_in[17];
  const float* g2ad = (const float*)d_in[18];
  const float* g2b  = (const float*)d_in[19];

  const int N = 64000, E = 2048000, B = 64, F = 40, C = 32, H1 = 8, D1 = 4;
  float* ws  = (float*)d_ws;
  float* xA  = ws;                  // N*32
  float* xB  = ws + 2048000;        // N*32
  float* xp  = ws + 4096000;        // N*32
  float* agg = ws + 6144000;        // N*32
  float* asb = ws + 8192000;        // N*8
  float* adb = ws + 8704000;        // N*8
  float* mb  = ws + 9216000;        // N*8
  float* sb  = ws + 9728000;        // N*8
  float* ysb = ws + 10240000;       // B*F*32
  float* red = ws + 10321920;       // 8
  float* b1p = ws + 10321928;       // 176 (padded ce1_b)
  float* b2p = ws + 10322104;       // 96  (padded ce2_b)
  __bf16* bfbase = (__bf16*)(ws + 10322208);   // 16-B aligned bf16 region
  __bf16* W1p = bfbase;             // 176x320
  __bf16* W2p = bfbase + 56320;     // 96x192
  __bf16* W3p = bfbase + 74752;     // 32x128
  __bf16* W4p = bfbase + 78848;     // 3x(32x32)
  __bf16* G1p = bfbase + 81920;     // 3x(32x32)
  __bf16* G2p = bfbase + 84992;     // 3x(32x32)

  // Pack weights (f32 -> zero-padded bf16) and biases (zero-padded f32).
  k_pack_w<<<221, 256, 0, stream>>>(ce1W, 172, 320, W1p, 176, 320);
  k_pack_w<<<72, 256, 0, stream>>>(ce2W, 86, 172, W2p, 96, 192);
  k_pack_w<<<16, 256, 0, stream>>>(lc0W, 32, 110, W3p, 32, 128);
  k_pack_w<<<12, 256, 0, stream>>>(lcW, 96, 32, W4p, 96, 32);
  k_pack_w<<<12, 256, 0, stream>>>(g1W, 96, 32, G1p, 96, 32);
  k_pack_w<<<12, 256, 0, stream>>>(g2W, 96, 32, G2p, 96, 32);
  k_pack_bias<<<1, 256, 0, stream>>>(ce1b, 172, b1p, 176);
  k_pack_bias<<<1, 256, 0, stream>>>(ce2b, 86, b2p, 96);

  // Embed nodes and fragments (WMMA MLP chain).
  k_embed<<<1000, 128, 0, stream>>>(node, N, W1p, b1p, W2p, b2p, W3p, lc0b,
                                    W4p, lcb, xA);
  k_embed<<<40, 128, 0, stream>>>(frag, B * F, W1p, b1p, W2p, b2p, W3p, lc0b,
                                  W4p, lcb, ysb);

  const int EP   = E + N;           // with self-loops
  const int tot1 = EP * H1;
  const int tot2 = EP;
  for (int l = 0; l < 3; ++l) {
    // ---- GAT1 (H=8, D=4, concat, elu) ----
    k_proj<<<1000, 128, 0, stream>>>(xA, N, G1p + l * C * C, xp);
    k_coef<8, 4><<<(N * H1 + 255) / 256, 256, 0, stream>>>(
        xp, N, g1as + l * C, g1ad + l * C, asb, adb);
    k_fill<<<2000, 256, 0, stream>>>(mb, -3.0e38f, N * H1);
    k_fill<<<2000, 256, 0, stream>>>(sb, 0.0f, N * H1);
    k_fill<<<8000, 256, 0, stream>>>(agg, 0.0f, N * C);
    k_edge_max<8><<<(tot1 + 255) / 256, 256, 0, stream>>>(ei, E, N, asb, adb, mb);
    k_edge_sum<8><<<(tot1 + 255) / 256, 256, 0, stream>>>(ei, E, N, asb, adb, mb, sb);
    k_edge_agg<8, 4><<<(tot1 + 255) / 256, 256, 0, stream>>>(ei, E, N, asb,
                                                             adb, mb, sb, xp, agg);
    k_bias_act<<<8000, 256, 0, stream>>>(agg, g1b + l * C, xB, N * C, 1);
    // ---- GAT2 (H=1, D=32, mean==identity, no act) ----
    k_proj<<<1000, 128, 0, stream>>>(xB, N, G2p + l * C * C, xp);
    k_coef<1, 32><<<(N + 255) / 256, 256, 0, stream>>>(
        xp, N, g2as + l * C, g2ad + l * C, asb, adb);
    k_fill<<<250, 256, 0, stream>>>(mb, -3.0e38f, N);
    k_fill<<<250, 256, 0, stream>>>(sb, 0.0f, N);
    k_fill<<<8000, 256, 0, stream>>>(agg, 0.0f, N * C);
    k_edge_max<1><<<(tot2 + 255) / 256, 256, 0, stream>>>(ei, E, N, asb, adb, mb);
    k_edge_sum<1><<<(tot2 + 255) / 256, 256, 0, stream>>>(ei, E, N, asb, adb, mb, sb);
    k_edge_agg<1, 32><<<(tot2 + 255) / 256, 256, 0, stream>>>(ei, E, N, asb,
                                                              adb, mb, sb, xp, agg);
    k_bias_act<<<8000, 256, 0, stream>>>(agg, g2b + l * C, xA, N * C, 0);
  }

  // Whole-tensor LN stats + per-graph scoring.
  k_fill<<<1, 256, 0, stream>>>(red, 0.0f, 8);
  k_redsum<<<1024, 256, 0, stream>>>(xA, N * C, red);
  k_redsum<<<64, 256, 0, stream>>>(ysb, B * F * C, red + 2);
  k_score<<<B, 256, 0, stream>>>(xA, ysb, red, (float*)d_out, N / B, F,
                                 (float)(N * C), (float)(B * F * C));
}
